// SSD_87892210745802
// MI455X (gfx1250) — matrix operand order repaired and checked by reading, not compiled, not run on hardware
//
#include <hip/hip_runtime.h>
#include <hip/hip_bf16.h>

typedef __attribute__((ext_vector_type(16))) _Float16 v16h;
typedef __attribute__((ext_vector_type(8)))  _Float16 v8h;
typedef __attribute__((ext_vector_type(8)))  float    v8f;

// ---------------- conversion kernels ----------------

__global__ void k_nchw_to_nhwc_f16(const float* __restrict__ x, _Float16* __restrict__ y,
                                   int N, int C, int H, int W) {
  size_t i = (size_t)blockIdx.x * blockDim.x + threadIdx.x;
  size_t tot = (size_t)N * C * H * W;
  if (i >= tot) return;
  int c = (int)(i % C); size_t t = i / C;
  int xw = (int)(t % W); t /= W;
  int yy = (int)(t % H); int n = (int)(t / H);
  y[i] = (_Float16)x[(((size_t)n * C + c) * H + yy) * W + xw];
}

// Pack OIHW f32 weights -> f16 [Opad][Kpad], K-order = (ky*Ks+kx)*Cin + ci, zero padded.
__global__ void k_wpack(const float* __restrict__ w, _Float16* __restrict__ y,
                        int Cout, int Cin, int Ks, int Opad, int Kpad) {
  size_t i = (size_t)blockIdx.x * blockDim.x + threadIdx.x;
  size_t tot = (size_t)Opad * Kpad;
  if (i >= tot) return;
  int kk = (int)(i % Kpad);
  int o  = (int)(i / Kpad);
  float v = 0.f;
  int Kdim = Cin * Ks * Ks;
  if (o < Cout && kk < Kdim) {
    int s  = kk / Cin;
    int ci = kk - s * Cin;
    int ky = s / Ks;
    int kx = s - ky * Ks;
    v = w[(((size_t)o * Cin + ci) * Ks + ky) * Ks + kx];
  }
  y[i] = (_Float16)v;
}

// ---------------- pooling / norm (NHWC f16) ----------------

__global__ void k_pool2(const _Float16* __restrict__ x, _Float16* __restrict__ y,
                        int N, int C, int Hin, int Win, int Hout, int Wout) {
  size_t i = (size_t)blockIdx.x * blockDim.x + threadIdx.x;
  size_t tot = (size_t)N * Hout * Wout * C;
  if (i >= tot) return;
  int c = (int)(i % C); size_t t = i / C;
  int ox = (int)(t % Wout); t /= Wout;
  int oy = (int)(t % Hout); int n = (int)(t / Hout);
  float m = -3.4e38f;
  for (int dy = 0; dy < 2; ++dy) {
    int iy = oy * 2 + dy; if (iy >= Hin) continue;
    for (int dx = 0; dx < 2; ++dx) {
      int ix = ox * 2 + dx; if (ix >= Win) continue;
      float v = (float)x[(((size_t)n * Hin + iy) * Win + ix) * C + c];
      m = v > m ? v : m;
    }
  }
  y[i] = (_Float16)m;
}

__global__ void k_pool3s1(const _Float16* __restrict__ x, _Float16* __restrict__ y,
                          int N, int C, int H, int W) {
  size_t i = (size_t)blockIdx.x * blockDim.x + threadIdx.x;
  size_t tot = (size_t)N * H * W * C;
  if (i >= tot) return;
  int c = (int)(i % C); size_t t = i / C;
  int ox = (int)(t % W); t /= W;
  int oy = (int)(t % H); int n = (int)(t / H);
  float m = -3.4e38f;
  for (int dy = -1; dy <= 1; ++dy) {
    int iy = oy + dy; if ((unsigned)iy >= (unsigned)H) continue;
    for (int dx = -1; dx <= 1; ++dx) {
      int ix = ox + dx; if ((unsigned)ix >= (unsigned)W) continue;
      float v = (float)x[(((size_t)n * H + iy) * W + ix) * C + c];
      m = v > m ? v : m;
    }
  }
  y[i] = (_Float16)m;
}

__global__ void k_l2norm(const _Float16* __restrict__ x, const float* __restrict__ sc,
                         _Float16* __restrict__ y, int NHW, int C) {
  int i = blockIdx.x * blockDim.x + threadIdx.x;
  if (i >= NHW) return;
  const _Float16* px = x + (size_t)i * C;
  _Float16* py = y + (size_t)i * C;
  float s = 0.f;
  for (int c = 0; c < C; ++c) { float v = (float)px[c]; s += v * v; }
  float inv = 1.f / (sqrtf(s) + 1e-10f);
  for (int c = 0; c < C; ++c) py[c] = (_Float16)((float)px[c] * inv * sc[c]);
}

// ---------------- implicit-GEMM conv via WMMA ----------------
// One wave computes a 64(M) x 32(N) block: 4 M-tiles x 2 N-tiles.
// Per 32-K step: 8 A-loads + 4 B-loads (b128) feed 8 v_wmma_f32_16x16x32_f16
// (1.5 loads/WMMA). Hot loop is division-free: outer loop over kernel taps
// (<=9), inner over 32-channel groups with purely additive addressing.

__global__ __launch_bounds__(256)
void k_conv_wmma(const _Float16* __restrict__ xh, const _Float16* __restrict__ wh,
                 const float* __restrict__ bias,
                 _Float16* __restrict__ yh, float* __restrict__ yhead,
                 int N, int Cin, int Hin, int Win,
                 int Cout, int Hout, int Wout,
                 int Ks, int stride, int pad, int dil,
                 int Kdim, int Kpad, int relu,
                 int head_dim, int head_npr, int anchor_off, int head_total) {
  const int NHW = N * Hout * Wout;
  const int Mt4 = (Cout + 63) >> 6;
  const int Nt2 = (NHW + 31) >> 5;
  const int wave = blockIdx.x * (blockDim.x >> 5) + ((int)threadIdx.x >> 5);
  if (wave >= Mt4 * Nt2) return;
  const int mt4 = wave / Nt2;          // waves in a block share mt4 -> A reuse in WGP$
  const int nt  = wave - mt4 * Nt2;
  const int lane = (int)threadIdx.x & 31;
  const int hh = lane >> 4;            // 16-lane half selects K sub-block
  const int lr = lane & 15;

  // two B columns per lane -> (batch, oy, ox)
  const int HW = Hout * Wout;
  const int col0 = nt * 32 + lr;
  const int col1 = col0 + 16;
  const int cc0 = col0 < NHW ? col0 : NHW - 1;
  const int cc1 = col1 < NHW ? col1 : NHW - 1;
  const int bN0 = cc0 / HW;       const int bN1 = cc1 / HW;
  const int re0 = cc0 - bN0 * HW; const int re1 = cc1 - bN1 * HW;
  const int oy0 = re0 / Wout;     const int oy1 = re1 / Wout;
  const int ox0 = re0 - oy0 * Wout; const int ox1 = re1 - oy1 * Wout;
  const int iy00 = oy0 * stride - pad; const int ix00 = ox0 * stride - pad;
  const int iy01 = oy1 * stride - pad; const int ix01 = ox1 * stride - pad;

  // 4 weight-row pointers (rows padded to Opad = multiple of 64, zero-filled)
  const _Float16* a0 = wh + (size_t)(mt4 * 64 + lr) * Kpad + hh * 16;
  const _Float16* a1 = a0 + (size_t)16 * Kpad;
  const _Float16* a2 = a1 + (size_t)16 * Kpad;
  const _Float16* a3 = a2 + (size_t)16 * Kpad;

  v8f acc[2][4] = {{{}, {}, {}, {}}, {{}, {}, {}, {}}};

  if ((Cin & 31) == 0) {
    // ---- vector path: every non-stem layer (Cin in {64..1024})
    const int KK = Ks * Ks;
    int kk = 0;                                  // = s*Cin + c0
    for (int s = 0; s < KK; ++s) {
      const int ky = s / Ks;
      const int kx = s - ky * Ks;
      const int iyA = iy00 + ky * dil, ixA = ix00 + kx * dil;
      const int iyB = iy01 + ky * dil, ixB = ix01 + kx * dil;
      const bool v0 = (unsigned)iyA < (unsigned)Hin && (unsigned)ixA < (unsigned)Win;
      const bool v1 = (unsigned)iyB < (unsigned)Hin && (unsigned)ixB < (unsigned)Win;
      const _Float16* bb0 =
          xh + (((size_t)(bN0 * Hin + (v0 ? iyA : 0))) * Win + (v0 ? ixA : 0)) * Cin + hh * 16;
      const _Float16* bb1 =
          xh + (((size_t)(bN1 * Hin + (v1 ? iyB : 0))) * Win + (v1 ? ixB : 0)) * Cin + hh * 16;
      for (int c0 = 0; c0 < Cin; c0 += 32, kk += 32) {
        v16h b0 = {}, b1 = {};
        if (v0) { const uint4* p = (const uint4*)(bb0 + c0); ((uint4*)&b0)[0] = p[0]; ((uint4*)&b0)[1] = p[1]; }
        if (v1) { const uint4* p = (const uint4*)(bb1 + c0); ((uint4*)&b1)[0] = p[0]; ((uint4*)&b1)[1] = p[1]; }
        v16h A0, A1, A2, A3;
        { const uint4* p = (const uint4*)(a0 + kk); ((uint4*)&A0)[0] = p[0]; ((uint4*)&A0)[1] = p[1]; }
        { const uint4* p = (const uint4*)(a1 + kk); ((uint4*)&A1)[0] = p[0]; ((uint4*)&A1)[1] = p[1]; }
        { const uint4* p = (const uint4*)(a2 + kk); ((uint4*)&A2)[0] = p[0]; ((uint4*)&A2)[1] = p[1]; }
        { const uint4* p = (const uint4*)(a3 + kk); ((uint4*)&A3)[0] = p[0]; ((uint4*)&A3)[1] = p[1]; }
        __builtin_prefetch(a0 + kk + 256, 0, 1);
        acc[0][0] = __builtin_amdgcn_wmma_f32_16x16x32_f16(false, A0, false, b0, (short)0, acc[0][0], false, false);
        acc[0][1] = __builtin_amdgcn_wmma_f32_16x16x32_f16(false, A1, false, b0, (short)0, acc[0][1], false, false);
        acc[0][2] = __builtin_amdgcn_wmma_f32_16x16x32_f16(false, A2, false, b0, (short)0, acc[0][2], false, false);
        acc[0][3] = __builtin_amdgcn_wmma_f32_16x16x32_f16(false, A3, false, b0, (short)0, acc[0][3], false, false);
        acc[1][0] = __builtin_amdgcn_wmma_f32_16x16x32_f16(false, A0, false, b1, (short)0, acc[1][0], false, false);
        acc[1][1] = __builtin_amdgcn_wmma_f32_16x16x32_f16(false, A1, false, b1, (short)0, acc[1][1], false, false);
        acc[1][2] = __builtin_amdgcn_wmma_f32_16x16x32_f16(false, A2, false, b1, (short)0, acc[1][2], false, false);
        acc[1][3] = __builtin_amdgcn_wmma_f32_16x16x32_f16(false, A3, false, b1, (short)0, acc[1][3], false, false);
      }
    }
  } else {
    // ---- stem path (Cin=3): scalar gather, Kpad=32 -> single step
    for (int k0 = 0; k0 < Kpad; k0 += 32) {
      v16h b0 = {}, b1 = {};
      const int kb = k0 + hh * 16;
      #pragma unroll
      for (int e = 0; e < 16; ++e) {
        const int k = kb + e;
        if (k < Kdim) {
          const int s  = k / Cin;
          const int ci = k - s * Cin;
          const int ky = s / Ks;
          const int kx = s - ky * Ks;
          const int iyA = iy00 + ky * dil, ixA = ix00 + kx * dil;
          const int iyB = iy01 + ky * dil, ixB = ix01 + kx * dil;
          if ((unsigned)iyA < (unsigned)Hin && (unsigned)ixA < (unsigned)Win)
            b0[e] = xh[(((size_t)(bN0 * Hin + iyA)) * Win + ixA) * Cin + ci];
          if ((unsigned)iyB < (unsigned)Hin && (unsigned)ixB < (unsigned)Win)
            b1[e] = xh[(((size_t)(bN1 * Hin + iyB)) * Win + ixB) * Cin + ci];
        }
      }
      v16h A0, A1, A2, A3;
      { const uint4* p = (const uint4*)(a0 + k0); ((uint4*)&A0)[0] = p[0]; ((uint4*)&A0)[1] = p[1]; }
      { const uint4* p = (const uint4*)(a1 + k0); ((uint4*)&A1)[0] = p[0]; ((uint4*)&A1)[1] = p[1]; }
      { const uint4* p = (const uint4*)(a2 + k0); ((uint4*)&A2)[0] = p[0]; ((uint4*)&A2)[1] = p[1]; }
      { const uint4* p = (const uint4*)(a3 + k0); ((uint4*)&A3)[0] = p[0]; ((uint4*)&A3)[1] = p[1]; }
      acc[0][0] = __builtin_amdgcn_wmma_f32_16x16x32_f16(false, A0, false, b0, (short)0, acc[0][0], false, false);
      acc[0][1] = __builtin_amdgcn_wmma_f32_16x16x32_f16(false, A1, false, b0, (short)0, acc[0][1], false, false);
      acc[0][2] = __builtin_amdgcn_wmma_f32_16x16x32_f16(false, A2, false, b0, (short)0, acc[0][2], false, false);
      acc[0][3] = __builtin_amdgcn_wmma_f32_16x16x32_f16(false, A3, false, b0, (short)0, acc[0][3], false, false);
      acc[1][0] = __builtin_amdgcn_wmma_f32_16x16x32_f16(false, A0, false, b1, (short)0, acc[1][0], false, false);
      acc[1][1] = __builtin_amdgcn_wmma_f32_16x16x32_f16(false, A1, false, b1, (short)0, acc[1][1], false, false);
      acc[1][2] = __builtin_amdgcn_wmma_f32_16x16x32_f16(false, A2, false, b1, (short)0, acc[1][2], false, false);
      acc[1][3] = __builtin_amdgcn_wmma_f32_16x16x32_f16(false, A3, false, b1, (short)0, acc[1][3], false, false);
    }
  }

  // ---- epilogue: bias (+ReLU) and store (both N-tiles)
  const bool vec_out = (head_dim == 0) && ((Cout & 63) == 0);
  #pragma unroll
  for (int t = 0; t < 2; ++t) {
    const int col = t ? col1 : col0;
    if (col >= NHW) continue;
    const int bN = t ? bN1 : bN0;
    const int oy = t ? oy1 : oy0;
    const int ox = t ? ox1 : ox0;
    const size_t obase = (((size_t)(bN * Hout + oy)) * Wout + ox) * (size_t)Cout;
    #pragma unroll
    for (int j = 0; j < 4; ++j) {
      const int Mb = mt4 * 64 + j * 16 + 8 * hh;   // C/D layout: VGPR v -> M = v + 8*half
      if (Mb < Cout) {
        if (vec_out) {                               // 8 contiguous channels, one b128 store
          v8h outv;
          #pragma unroll
          for (int v = 0; v < 8; ++v) {
            float val = acc[t][j][v] + bias[Mb + v];
            if (relu) val = fmaxf(val, 0.f);
            outv[v] = (_Float16)val;
          }
          *(v8h*)(yh + obase + Mb) = outv;
        } else if (head_dim == 0) {
          #pragma unroll
          for (int v = 0; v < 8; ++v) {
            const int M = Mb + v;
            if (M < Cout) {
              float val = acc[t][j][v] + bias[M];
              if (relu) val = fmaxf(val, 0.f);
              yh[obase + M] = (_Float16)val;
            }
          }
        } else {
          #pragma unroll
          for (int v = 0; v < 8; ++v) {
            const int M = Mb + v;
            if (M < Cout) {
              const float val = acc[t][j][v] + bias[M];
              const int pr = M / head_dim;
              const int d  = M - pr * head_dim;
              const int anchor = anchor_off + (oy * Wout + ox) * head_npr + pr;
              yhead[((size_t)bN * head_total + anchor) * (size_t)head_dim + d] = val;
            }
          }
        }
      }
    }
  }
}

// ---------------- host orchestration ----------------

extern "C" void kernel_launch(void* const* d_in, const int* in_sizes, int n_in,
                              void* d_out, int out_size, void* d_ws, size_t ws_size,
                              hipStream_t stream) {
  (void)in_sizes; (void)n_in; (void)out_size; (void)ws_size;
  const int NB = 8;

  const float* x_in = (const float*)d_in[0];
  const float* vw[15]; const float* vb[15];
  for (int i = 0; i < 15; ++i) { vw[i] = (const float*)d_in[1 + i]; vb[i] = (const float*)d_in[16 + i]; }
  const float* l2s = (const float*)d_in[31];
  const float* ew[8]; const float* eb[8];
  for (int i = 0; i < 8; ++i) { ew[i] = (const float*)d_in[32 + i]; eb[i] = (const float*)d_in[40 + i]; }
  const float* lw[6]; const float* lb[6]; const float* cw2[6]; const float* cb2[6];
  for (int i = 0; i < 6; ++i) {
    lw[i]  = (const float*)d_in[48 + i]; lb[i]  = (const float*)d_in[54 + i];
    cw2[i] = (const float*)d_in[60 + i]; cb2[i] = (const float*)d_in[66 + i];
  }

  char* wsb = (char*)d_ws; size_t off = 0;
  auto balloc = [&](size_t bytes) -> char* {
    char* p = wsb + off; off += (bytes + 255) & ~(size_t)255; return p;
  };

  struct Conv { const float* w; const float* b; int Cin, Cout, K, Kdim, Kpad, Opad; _Float16* wh; };
  Conv conv[35]; int nconv = 0;
  auto addc = [&](const float* w, const float* b, int ci, int co, int k) -> int {
    Conv c; c.w = w; c.b = b; c.Cin = ci; c.Cout = co; c.K = k;
    c.Kdim = ci * k * k; c.Kpad = (c.Kdim + 31) & ~31; c.Opad = (co + 63) & ~63;
    c.wh = (_Float16*)balloc((size_t)c.Opad * c.Kpad * sizeof(_Float16));
    conv[nconv] = c; return nconv++;
  };

  const int vci[15] = {3,64,64,128,128,256,256,256,512,512,512,512,512,512,1024};
  const int vco[15] = {64,64,128,128,256,256,256,512,512,512,512,512,512,1024,1024};
  const int vkk[15] = {3,3,3,3,3,3,3,3,3,3,3,3,3,3,1};
  int idv[15]; for (int i = 0; i < 15; ++i) idv[i] = addc(vw[i], vb[i], vci[i], vco[i], vkk[i]);
  const int eci[8] = {1024,256,512,128,256,128,256,128};
  const int eco[8] = {256,512,128,256,128,256,128,256};
  const int ekk[8] = {1,3,1,3,1,3,1,3};
  int ide[8]; for (int i = 0; i < 8; ++i) ide[i] = addc(ew[i], eb[i], eci[i], eco[i], ekk[i]);
  const int fch[6] = {512,1024,512,256,256,256};
  const int npr[6] = {4,6,6,6,4,4};
  int idl[6], idc[6];
  for (int i = 0; i < 6; ++i) idl[i] = addc(lw[i],  lb[i],  fch[i], npr[i] * 4,  3);
  for (int i = 0; i < 6; ++i) idc[i] = addc(cw2[i], cb2[i], fch[i], npr[i] * 21, 3);

  // pack all weights to f16
  for (int i = 0; i < nconv; ++i) {
    Conv& c = conv[i];
    size_t tot = (size_t)c.Opad * c.Kpad;
    k_wpack<<<(unsigned)((tot + 255) / 256), 256, 0, stream>>>(c.w, c.wh, c.Cout, c.Cin, c.K, c.Opad, c.Kpad);
  }

  // activation buffers (NHWC f16)
  const size_t maxh = (size_t)NB * 300 * 300 * 64;
  _Float16* actA  = (_Float16*)balloc(maxh * 2);
  _Float16* actB  = (_Float16*)balloc(maxh * 2);
  _Float16* feat0 = (_Float16*)balloc((size_t)NB * 38 * 38 * 512 * 2);
  _Float16* feat1 = (_Float16*)balloc((size_t)NB * 19 * 19 * 1024 * 2);
  _Float16* feat2 = (_Float16*)balloc((size_t)NB * 10 * 10 * 512 * 2);
  _Float16* feat3 = (_Float16*)balloc((size_t)NB * 5 * 5 * 256 * 2);
  _Float16* feat4 = (_Float16*)balloc((size_t)NB * 3 * 3 * 256 * 2);
  _Float16* feat5 = (_Float16*)balloc((size_t)NB * 1 * 1 * 256 * 2);

  _Float16* cur = actA; _Float16* nxt = actB;
  auto swp = [&]() { _Float16* t = cur; cur = nxt; nxt = t; };

  { // input NCHW f32 -> NHWC f16
    size_t tot = (size_t)NB * 3 * 300 * 300;
    k_nchw_to_nhwc_f16<<<(unsigned)((tot + 255) / 256), 256, 0, stream>>>(x_in, cur, NB, 3, 300, 300);
  }

  auto run_conv = [&](int id, const _Float16* src, _Float16* dst, float* hdst,
                      int Hin, int stride, int pad, int dil, int relu,
                      int hd, int hn, int aoff) {
    Conv& c = conv[id];
    int Hout = (Hin + 2 * pad - dil * (c.K - 1) - 1) / stride + 1;
    int NHW = NB * Hout * Hout;
    int Mt4 = (c.Cout + 63) / 64, Nt2 = (NHW + 31) / 32;
    long waves = (long)Mt4 * Nt2;
    unsigned blocks = (unsigned)((waves + 7) / 8);
    k_conv_wmma<<<blocks, 256, 0, stream>>>(src, c.wh, c.b, dst, hdst,
        NB, c.Cin, Hin, Hin, c.Cout, Hout, Hout, c.K, stride, pad, dil,
        c.Kdim, c.Kpad, relu, hd, hn, aoff, 8732);
  };
  auto run_pool2 = [&](int C, int Hin, int Hout) {
    size_t tot = (size_t)NB * Hout * Hout * C;
    k_pool2<<<(unsigned)((tot + 255) / 256), 256, 0, stream>>>(cur, nxt, NB, C, Hin, Hin, Hout, Hout);
    swp();
  };

  // VGG trunk
  run_conv(idv[0],  cur, nxt, nullptr, 300, 1, 1, 1, 1, 0, 0, 0); swp();
  run_conv(idv[1],  cur, nxt, nullptr, 300, 1, 1, 1, 1, 0, 0, 0); swp();
  run_pool2(64, 300, 150);
  run_conv(idv[2],  cur, nxt, nullptr, 150, 1, 1, 1, 1, 0, 0, 0); swp();
  run_conv(idv[3],  cur, nxt, nullptr, 150, 1, 1, 1, 1, 0, 0, 0); swp();
  run_pool2(128, 150, 75);
  run_conv(idv[4],  cur, nxt, nullptr, 75, 1, 1, 1, 1, 0, 0, 0); swp();
  run_conv(idv[5],  cur, nxt, nullptr, 75, 1, 1, 1, 1, 0, 0, 0); swp();
  run_conv(idv[6],  cur, nxt, nullptr, 75, 1, 1, 1, 1, 0, 0, 0); swp();
  run_pool2(256, 75, 38);                                        // ceil mode
  run_conv(idv[7],  cur, nxt, nullptr, 38, 1, 1, 1, 1, 0, 0, 0); swp();
  run_conv(idv[8],  cur, nxt, nullptr, 38, 1, 1, 1, 1, 0, 0, 0); swp();
  run_conv(idv[9],  cur, nxt, nullptr, 38, 1, 1, 1, 1, 0, 0, 0); swp();  // conv4_3
  k_l2norm<<<(NB * 38 * 38 + 255) / 256, 256, 0, stream>>>(cur, l2s, feat0, NB * 38 * 38, 512);
  run_pool2(512, 38, 19);
  run_conv(idv[10], cur, nxt, nullptr, 19, 1, 1, 1, 1, 0, 0, 0); swp();
  run_conv(idv[11], cur, nxt, nullptr, 19, 1, 1, 1, 1, 0, 0, 0); swp();
  run_conv(idv[12], cur, nxt, nullptr, 19, 1, 1, 1, 1, 0, 0, 0); swp();
  { size_t tot = (size_t)NB * 19 * 19 * 512;                      // pool5 3x3/1
    k_pool3s1<<<(unsigned)((tot + 255) / 256), 256, 0, stream>>>(cur, nxt, NB, 512, 19, 19); swp(); }
  run_conv(idv[13], cur, nxt, nullptr, 19, 1, 6, 6, 1, 0, 0, 0); swp();  // conv6 dilated
  run_conv(idv[14], cur, feat1, nullptr, 19, 1, 0, 1, 1, 0, 0, 0);       // conv7 -> feat1

  // Extras
  _Float16* t0 = actA;
  run_conv(ide[0], feat1, t0,    nullptr, 19, 1, 0, 1, 1, 0, 0, 0);
  run_conv(ide[1], t0,    feat2, nullptr, 19, 2, 1, 1, 1, 0, 0, 0);
  run_conv(ide[2], feat2, t0,    nullptr, 10, 1, 0, 1, 1, 0, 0, 0);
  run_conv(ide[3], t0,    feat3, nullptr, 10, 2, 1, 1, 1, 0, 0, 0);
  run_conv(ide[4], feat3, t0,    nullptr, 5, 1, 0, 1, 1, 0, 0, 0);
  run_conv(ide[5], t0,    feat4, nullptr, 5, 1, 0, 1, 1, 0, 0, 0);
  run_conv(ide[6], feat4, t0,    nullptr, 3, 1, 0, 1, 1, 0, 0, 0);
  run_conv(ide[7], t0,    feat5, nullptr, 3, 1, 0, 1, 1, 0, 0, 0);

  // Heads -> d_out (locs then confs, anchor-major)
  float* loc  = (float*)d_out;
  float* conf = (float*)d_out + (size_t)NB * 8732 * 4;
  _Float16* feats[6] = {feat0, feat1, feat2, feat3, feat4, feat5};
  const int fH[6]    = {38, 19, 10, 5, 3, 1};
  const int aoff[6]  = {0, 5776, 7942, 8542, 8692, 8728};
  for (int i = 0; i < 6; ++i) {
    run_conv(idl[i], feats[i], nullptr, loc,  fH[i], 1, 1, 1, 0,  4, npr[i], aoff[i]);
    run_conv(idc[i], feats[i], nullptr, conf, fH[i], 1, 1, 1, 0, 21, npr[i], aoff[i]);
  }
}
